// monkey_switch_group_QKV_cromul_22539988369548
// MI455X (gfx1250) — matrix operand order repaired and checked by reading, not compiled, not run on hardware
//
#include <hip/hip_runtime.h>
#include <hip/hip_bf16.h>
#include <math.h>

#define B_    256
#define INC   2304
#define HID_  512
#define HW_   49
#define NCHUNK 72              // 2304 / 32
#define OUTSZ (B_ * HID_ * HW_)   // 6422528 per branch
#define SCALE_ 0.04419417382415922f  // 1/sqrt(512)
#define CELLS 102              // 81 real cells + 21 always-zero cells for OOB rows
#define CSTR  40               // padded cell stride (ushorts): 80 B -> conflict-free banks

typedef __attribute__((ext_vector_type(16))) __bf16 v16bf;
typedef __attribute__((ext_vector_type(8)))  float  v8f;

union BFrag { v16bf v; uint4 q[2]; };

__device__ __forceinline__ unsigned short f2bf(float f) {
    union { float f; unsigned int u; } c; c.f = f;
    unsigned int u = c.u;
    u += 0x7FFFu + ((u >> 16) & 1u);   // round-to-nearest-even
    return (unsigned short)(u >> 16);
}

__device__ __forceinline__ float wsum(float v) {
    #pragma unroll
    for (int o = 16; o; o >>= 1) v += __shfl_xor(v, o, 32);
    return v;
}
__device__ __forceinline__ float wmaxr(float v) {
    #pragma unroll
    for (int o = 16; o; o >>= 1) v = fmaxf(v, __shfl_xor(v, o, 32));
    return v;
}

// ---------------------------------------------------------------------------
// K0: pack conv_w (512,2304,3,3) fp32 -> bf16, swizzled to per-lane WMMA
// B-fragment order: wB[(cc*9+t)*32+ctg][lane][16] with k_local = half*16 + i.
// ---------------------------------------------------------------------------
__global__ void pack_w_kernel(const float* __restrict__ w,
                              unsigned short* __restrict__ wB) {
    const int total = NCHUNK * 9 * 32 * 32 * 16;   // 10,616,832
    for (int e = blockIdx.x * blockDim.x + threadIdx.x; e < total;
         e += gridDim.x * blockDim.x) {
        int i    = e & 15;
        int lane = (e >> 4) & 31;
        int ctg  = (e >> 9) & 31;
        int ktg  = e >> 14;
        int cc = ktg / 9, t = ktg % 9;
        int hl = lane >> 4, n = lane & 15;
        int co = ctg * 16 + n;
        int ci = cc * 32 + hl * 16 + i;
        int kh = t / 3, kw = t % 3;
        wB[e] = f2bf(w[((co * INC + ci) * 3 + kh) * 3 + kw]);
    }
}

// ---------------------------------------------------------------------------
// K1: implicit-GEMM conv via v_wmma_f32_16x16x32_bf16.
// Block = 2 batches x 256 output channels, 8 waves. Each wave owns TWO co
// tiles (so each A fragment from LDS feeds 2 WMMAs: 1 ds_load_b128 per WMMA)
// and iterates all 8 M-tiles (2 batches x 4 position tiles) -> 16 acc tiles.
// Weights are loaded from global exactly once per block per K-step.
// ---------------------------------------------------------------------------
__global__ void __launch_bounds__(256) conv_wmma_kernel(
    const float* __restrict__ x,              // bbox_feat [B][2304][49]
    const unsigned short* __restrict__ wB,    // packed weights
    float* __restrict__ feat)                 // [B][49][512]
{
    __shared__ unsigned short ldsx[2][CELLS * CSTR];  // 16320 B

    const int bb  = blockIdx.x >> 1;          // batch pair
    const int cog = blockIdx.x & 1;           // co base = cog*256
    const int b0  = bb * 2;
    const int tid = threadIdx.x;
    const int wv  = tid >> 5;                 // wave 0..7
    const int lane = tid & 31;
    const int hl   = lane >> 4;
    const int m    = lane & 15;
    const int ct0  = cog * 16 + wv * 2;       // this wave's first co-tile (of 32)

    // per-M-tile lane geometry: base cell in un-haloed coords; 81 = zero region
    int cbase[4];
    #pragma unroll
    for (int q = 0; q < 4; ++q) {
        int p = q * 16 + m;
        cbase[q] = (p < HW_) ? ((p / 7) * 9 + (p % 7)) : 81;
    }

    for (int i = tid; i < CELLS * CSTR; i += 256) { ldsx[0][i] = 0; ldsx[1][i] = 0; }
    __syncthreads();

    v8f acc[16] = {};
    const uint4* wb4 = (const uint4*)wB;

    for (int cc = 0; cc < NCHUNK; ++cc) {
        // stage 2 batches x 32 channels x 49 positions as bf16 (haloed grid)
        for (int idx = tid; idx < 2 * 32 * HW_; idx += 256) {
            int bi = idx / (32 * HW_);
            int r  = idx - bi * 32 * HW_;
            int ci = r / HW_;
            int pp = r - ci * HW_;
            float v = x[((long long)(b0 + bi) * INC + cc * 32 + ci) * HW_ + pp];
            int yy = pp / 7 + 1, xx = pp % 7 + 1;
            ldsx[bi][(yy * 9 + xx) * CSTR + ci] = f2bf(v);
        }
        __syncthreads();

        #pragma unroll
        for (int t = 0; t < 9; ++t) {
            const int kh = t / 3, kw = t % 3;
            const int coff = kh * 9 + kw;     // tap offset in cell space
            const int ktg = cc * 9 + t;
            // this wave's two B fragments: 32 contiguous bytes per lane each
            const size_t wbase = ((size_t)(ktg * 32 + ct0) * 32 + lane) * 2;
            BFrag bf0, bf1;
            bf0.q[0] = wb4[wbase];      bf0.q[1] = wb4[wbase + 1];
            bf1.q[0] = wb4[wbase + 64]; bf1.q[1] = wb4[wbase + 65];
            // prefetch next K-step's weight line (global_prefetch_b8)
            __builtin_prefetch((const void*)((const char*)(wb4 + wbase) + 32768), 0, 1);

            #pragma unroll
            for (int q8 = 0; q8 < 8; ++q8) {
                const int bi = q8 >> 2, mt = q8 & 3;
                const int cell = cbase[mt] + coff;
                BFrag a;
                a.q[0] = *(const uint4*)&ldsx[bi][cell * CSTR + hl * 16];      // K 0..7 (+hl*16)
                a.q[1] = *(const uint4*)&ldsx[bi][cell * CSTR + hl * 16 + 8];  // K 8..15 (+hl*16)
                acc[q8 * 2] = __builtin_amdgcn_wmma_f32_16x16x32_bf16(
                    false, a.v, false, bf0.v, (short)0, acc[q8 * 2], false, false);
                acc[q8 * 2 + 1] = __builtin_amdgcn_wmma_f32_16x16x32_bf16(
                    false, a.v, false, bf1.v, (short)0, acc[q8 * 2 + 1], false, false);
            }
        }
        __syncthreads();
    }

    // store D tiles: lanes 0-15 -> M=r, N=lane; lanes 16-31 -> M=8+r, N=lane-16
    #pragma unroll
    for (int q8 = 0; q8 < 8; ++q8) {
        const int b = b0 + (q8 >> 2);
        #pragma unroll
        for (int c2 = 0; c2 < 2; ++c2) {
            const int co = (ct0 + c2) * 16 + (lane & 15);
            #pragma unroll
            for (int r = 0; r < 8; ++r) {
                int pp = (q8 & 3) * 16 + r + hl * 8;
                if (pp < HW_) {
                    feat[((long long)b * HW_ + pp) * HID_ + co] = acc[q8 * 2 + c2][r];
                }
            }
        }
    }
}

// ---------------------------------------------------------------------------
// K2: q[b,d] = st[b] . q1_w[d,:]   (st = [status(2), rois[1:5]])
// ---------------------------------------------------------------------------
__global__ void q_kernel(const float* __restrict__ status,
                         const float* __restrict__ rois,
                         const float* __restrict__ q1w,
                         float* __restrict__ q) {
    int idx = blockIdx.x * blockDim.x + threadIdx.x;
    if (idx >= B_ * HID_) return;
    int b = idx / HID_, d = idx - b * HID_;
    const float* wr = q1w + d * 6;
    float v = status[b * 2] * wr[0] + status[b * 2 + 1] * wr[1];
    #pragma unroll
    for (int i = 0; i < 4; ++i) v += rois[b * 5 + 1 + i] * wr[2 + i];
    q[idx] = v;
}

// ---------------------------------------------------------------------------
// K3: qk[j][b][c] = sum_d q[b,d] * kj_w[d,c]   (so logits = feat . qk)
// ---------------------------------------------------------------------------
__global__ void __launch_bounds__(256) qk_kernel(
    const float* __restrict__ q,
    const float* __restrict__ k1w, const float* __restrict__ k2w,
    float* __restrict__ qk) {
    __shared__ float qs[HID_];
    int j = blockIdx.x >> 8;
    int b = blockIdx.x & 255;
    const float* kw = j ? k2w : k1w;
    int tid = threadIdx.x;
    qs[tid] = q[b * HID_ + tid];
    qs[tid + 256] = q[b * HID_ + tid + 256];
    __syncthreads();
    #pragma unroll
    for (int h = 0; h < 2; ++h) {
        int c = tid + h * 256;
        float acc = 0.f;
        for (int d = 0; d < HID_; ++d) acc += qs[d] * kw[d * HID_ + c];
        qk[(j * B_ + b) * HID_ + c] = acc;
    }
}

// ---------------------------------------------------------------------------
// K4: WcT[j][c][d] = sum_e pj_w[d,e] * vj_w[e,c]  (= (p_w @ v_w)^T, stored
// transposed so the per-batch matvec reads are coalesced in d)
// ---------------------------------------------------------------------------
__global__ void __launch_bounds__(256) wct_kernel(
    const float* __restrict__ v1w, const float* __restrict__ v2w,
    const float* __restrict__ p1w, const float* __restrict__ p2w,
    float* __restrict__ wct) {
    __shared__ float As[16][17];   // A[c_local][e_local] = v_w[e,c]
    __shared__ float Bs[16][17];   // B[e_local][d_local] = p_w[d,e]
    int j = blockIdx.z;
    const float* vw = j ? v2w : v1w;
    const float* pw = j ? p2w : p1w;
    int tx = threadIdx.x, ty = threadIdx.y;
    int d0 = blockIdx.x * 16, c0 = blockIdx.y * 16;
    float acc = 0.f;
    for (int e0 = 0; e0 < HID_; e0 += 16) {
        As[tx][ty] = vw[(e0 + ty) * HID_ + c0 + tx];   // coalesced in tx
        Bs[tx][ty] = pw[(d0 + ty) * HID_ + e0 + tx];   // coalesced in tx
        __syncthreads();
        #pragma unroll
        for (int k = 0; k < 16; ++k) acc += As[ty][k] * Bs[k][tx];
        __syncthreads();
    }
    wct[((j * HID_) + c0 + ty) * HID_ + d0 + tx] = acc;
}

// ---------------------------------------------------------------------------
// K5: per-batch fused attention + residual for both branches.
// ---------------------------------------------------------------------------
__global__ void __launch_bounds__(256) attn_kernel(
    const float* __restrict__ feat,   // [B][49][512]
    const float* __restrict__ qk,     // [2][B][512]
    const float* __restrict__ wct,    // [2][512][512] (c,d)
    const float* __restrict__ b1, const float* __restrict__ b2,
    float* __restrict__ out) {
    __shared__ float qk_s[2][HID_];
    __shared__ float att_s[2][64];
    __shared__ float wf_s[2][HID_];
    __shared__ float vt_s[2][HID_];
    __shared__ float tr[HW_ * 33];

    const int b = blockIdx.x;
    const int tid = threadIdx.x;
    const int wave = tid >> 5, lane = tid & 31;
    const float* fb = feat + (long long)b * HW_ * HID_;

    #pragma unroll
    for (int j = 0; j < 2; ++j) {
        qk_s[j][tid]       = qk[(j * B_ + b) * HID_ + tid] * SCALE_;
        qk_s[j][tid + 256] = qk[(j * B_ + b) * HID_ + tid + 256] * SCALE_;
    }
    __syncthreads();

    // logits[j][n] = feat[b,n,:] . qk_s[j]
    for (int n = wave; n < HW_; n += 8) {
        float p1 = 0.f, p2 = 0.f;
        for (int c = lane; c < HID_; c += 32) {
            float f = fb[n * HID_ + c];
            p1 += f * qk_s[0][c];
            p2 += f * qk_s[1][c];
        }
        p1 = wsum(p1); p2 = wsum(p2);
        if (lane == 0) { att_s[0][n] = p1; att_s[1][n] = p2; }
    }
    __syncthreads();

    // softmax over 49 (wave 0 -> branch 0, wave 1 -> branch 1)
    if (tid < 64) {
        int j = wave;
        float l0 = (lane < HW_) ? att_s[j][lane] : -1e30f;
        float l1 = (lane + 32 < HW_) ? att_s[j][lane + 32] : -1e30f;
        float mx = wmaxr(fmaxf(l0, l1));
        float e0 = (lane < HW_) ? __expf(l0 - mx) : 0.f;
        float e1 = (lane + 32 < HW_) ? __expf(l1 - mx) : 0.f;
        float s = wsum(e0 + e1);
        float inv = 1.0f / s;
        if (lane < HW_) att_s[j][lane] = e0 * inv;
        if (lane + 32 < HW_) att_s[j][lane + 32] = e1 * inv;
    }
    __syncthreads();

    // wfeat[j][c] = sum_n att[j][n] * feat[b,n,c]
    #pragma unroll
    for (int h = 0; h < 2; ++h) {
        int c = tid + h * 256;
        float a1 = 0.f, a2 = 0.f;
        for (int n = 0; n < HW_; ++n) {
            float f = fb[n * HID_ + c];
            a1 += att_s[0][n] * f;
            a2 += att_s[1][n] * f;
        }
        wf_s[0][c] = a1; wf_s[1][c] = a2;
    }
    __syncthreads();

    // virt[j][d] = bias[d] + sum_c WcT[j][c][d] * wfeat[j][c]
    #pragma unroll
    for (int h = 0; h < 2; ++h) {
        int d = tid + h * 256;
        float a1 = b1[d], a2 = b2[d];
        for (int c = 0; c < HID_; ++c) {
            a1 += wct[(0 * HID_ + c) * HID_ + d] * wf_s[0][c];
            a2 += wct[(1 * HID_ + c) * HID_ + d] * wf_s[1][c];
        }
        vt_s[0][d] = a1; vt_s[1][d] = a2;
    }

    // out[j][b][co][p] = feat[b][p][co] + virt[j][co]  (LDS transpose chunks)
    for (int co0 = 0; co0 < HID_; co0 += 32) {
        __syncthreads();
        for (int i = tid; i < HW_ * 32; i += 256) {
            int pp = i >> 5, cl = i & 31;
            tr[pp * 33 + cl] = fb[pp * HID_ + co0 + cl];   // coalesced read
        }
        __syncthreads();
        for (int i = tid; i < 32 * HW_; i += 256) {
            int cl = i / HW_, pp = i - cl * HW_;
            int co = co0 + cl;
            float base = tr[pp * 33 + cl];
            long long o = ((long long)b * HID_ + co) * HW_ + pp;  // coalesced write
            out[o]         = base + vt_s[0][co];
            out[OUTSZ + o] = base + vt_s[1][co];
        }
    }
}

// ---------------------------------------------------------------------------
extern "C" void kernel_launch(void* const* d_in, const int* in_sizes, int n_in,
                              void* d_out, int out_size, void* d_ws, size_t ws_size,
                              hipStream_t stream) {
    const float* status = (const float*)d_in[0];
    const float* rois   = (const float*)d_in[1];
    const float* bbox   = (const float*)d_in[2];
    const float* convw  = (const float*)d_in[3];
    const float* q1w    = (const float*)d_in[4];
    const float* k1w    = (const float*)d_in[5];
    const float* k2w    = (const float*)d_in[6];
    const float* v1w    = (const float*)d_in[7];
    const float* v2w    = (const float*)d_in[8];
    const float* p1w    = (const float*)d_in[9];
    const float* p1b    = (const float*)d_in[10];
    const float* p2w    = (const float*)d_in[11];
    const float* p2b    = (const float*)d_in[12];
    float* out = (float*)d_out;

    char* ws = (char*)d_ws;
    unsigned short* wB = (unsigned short*)ws;               //  21,233,664 B
    float* feat = (float*)(ws + 21233664);                  //  25,690,112 B
    float* q    = (float*)(ws + 21233664 + 25690112);       //     524,288 B
    float* qk   = (float*)(ws + 21233664 + 25690112 + 524288);      // 1 MB
    float* wct  = (float*)(ws + 21233664 + 25690112 + 524288 + 1048576); // 2 MB

    pack_w_kernel<<<4096, 256, 0, stream>>>(convw, wB);
    conv_wmma_kernel<<<(B_ / 2) * 2, 256, 0, stream>>>(bbox, wB, feat);
    q_kernel<<<(B_ * HID_ + 255) / 256, 256, 0, stream>>>(status, rois, q1w, q);
    qk_kernel<<<2 * B_, 256, 0, stream>>>(q, k1w, k2w, qk);
    wct_kernel<<<dim3(32, 32, 2), dim3(16, 16), 0, stream>>>(v1w, v2w, p1w, p2w, wct);
    attn_kernel<<<B_, 256, 0, stream>>>(feat, qk, wct, p1b, p2b, out);
}
